// Text2Lstm_58677843198151
// MI455X (gfx1250) — matrix operand order repaired and checked by reading, not compiled, not run on hardware
//
#include <hip/hip_runtime.h>
#include <math.h>

typedef _Float16 v16h __attribute__((ext_vector_type(16)));
typedef float    v8f  __attribute__((ext_vector_type(8)));

#define HV   100          // hidden
#define GN   400          // 4*H gate width
#define BB   64           // batch
#define TT   512          // seq len
#define DK0  320          // 300 padded to mult of 32
#define DK1  224          // 200 padded to mult of 32
#define KH   128          // 100 padded to mult of 32
#define ROWS (BB*TT)      // 32768

// ---------------- WMMA fragment loaders (CDNA5 16x16x32 f16 layouts) -------

// A tile 16x32 (MxK): lanes 0-15 hold row M=lane, K {k0..k0+7, k0+16..k0+23};
// lanes 16-31 hold row M=lane-16, K {k0+8..k0+15, k0+24..k0+31}.
template <typename P>
__device__ inline v16h load_afrag(const P* src, int stride, int m0, int k0, int l) {
    int m  = m0 + (l & 15);
    int kh = k0 + ((l >> 4) << 3);
    const P* p = src + m * stride + kh;
    v16h a;
#pragma unroll
    for (int i = 0; i < 8; i++) a[i] = (_Float16)p[i];
#pragma unroll
    for (int i = 0; i < 8; i++) a[8 + i] = (_Float16)p[16 + i];
    return a;
}

// B tile 32x16 (KxN) sourced from row-major W[N,K] (i.e. computing A @ W^T):
// lanes 0-15: col N=lane, K {k0..k0+7, k0+16..k0+23}; lanes 16-31: col N=lane-16, K {k0+8..k0+15, k0+24..k0+31}.
__device__ inline v16h load_bfrag(const _Float16* w, int stride, int n0, int k0, int l) {
    int n  = n0 + (l & 15);
    int kb = k0 + ((l >> 4) << 3);
    const _Float16* p = w + n * stride + kb;
    v16h b;
#pragma unroll
    for (int i = 0; i < 16; i++) b[i] = p[(i & 7) + ((i >> 3) << 4)];
    return b;
}

__device__ inline float sigm(float x) { return 1.0f / (1.0f + expf(-x)); }

// Work-group barrier with split counters: only wait on LDS (DScnt) so that
// outstanding global loads (xp prefetch) stay in flight across the barrier.
__device__ inline void wg_barrier_lds() {
    asm volatile("s_wait_dscnt 0x0\n\t"
                 "s_barrier_signal -1\n\t"
                 "s_barrier_wait -1" ::: "memory");
}

// ---------------- prep kernels ---------------------------------------------

// dst[r, 0..dc) = (c < sc) ? (f16)src[r, c] : 0   ; dst is [rows, dc] f16
__global__ void cvt_pad_kernel(const float* __restrict__ src, _Float16* __restrict__ dst,
                               int rows, int sc, int dc) {
    int idx = blockIdx.x * blockDim.x + threadIdx.x;
    int total = rows * dc;
    if (idx >= total) return;
    int r = idx / dc, c = idx - r * dc;
    dst[idx] = (c < sc) ? (_Float16)src[r * sc + c] : (_Float16)0.0f;
}

// bias sum: out[i] = a[i] + b[i], i < 400
__global__ void bias_sum_kernel(const float* __restrict__ a, const float* __restrict__ b,
                                float* __restrict__ out) {
    int i = blockIdx.x * blockDim.x + threadIdx.x;
    if (i < GN) out[i] = a[i] + b[i];
}

// embedding gather into f16 [T,B,DK0]: row r = t*64+b
__global__ void embed_kernel(const int* __restrict__ x, const float* __restrict__ emb,
                             _Float16* __restrict__ xe) {
    int r = blockIdx.x;           // t*64 + b
    int d = threadIdx.x;          // 0..319
    int t = r >> 6, b = r & 63;
    int idx = x[b * TT + t];
    xe[(size_t)r * DK0 + d] = (d < 300) ? (_Float16)emb[(size_t)idx * 300 + d] : (_Float16)0.0f;
}

// ---------------- input-projection GEMM ------------------------------------
// out[r, n] = sum_k A[r,k] * W[n,k] + bias[n] ; A [ROWS,SK] f16, W [400,SK] f16,
// out [ROWS,400] f32. block = 256 (8 waves), wave -> one 64x16 tile.
__global__ void proj_gemm_kernel(const _Float16* __restrict__ A,
                                 const _Float16* __restrict__ W0, const _Float16* __restrict__ W1,
                                 const float* __restrict__ bias0, const float* __restrict__ bias1,
                                 float* __restrict__ out0, float* __restrict__ out1,
                                 int SK) {
    const _Float16* W   = blockIdx.z ? W1 : W0;
    const float*    bia = blockIdx.z ? bias1 : bias0;
    float*          out = blockIdx.z ? out1 : out0;

    int w = threadIdx.x >> 5;
    int l = threadIdx.x & 31;
    int nt = blockIdx.x * 8 + w;        // N tile 0..24
    if (nt >= 25) return;               // wave-uniform
    int n0 = nt * 16;
    int m0 = blockIdx.y * 64;

    v8f acc[4];
#pragma unroll
    for (int mt = 0; mt < 4; mt++) acc[mt] = (v8f)(0.0f);

    int nk = SK >> 5;
    for (int kt = 0; kt < nk; kt++) {
        v16h bf = load_bfrag(W, SK, n0, kt * 32, l);
#pragma unroll
        for (int mt = 0; mt < 4; mt++) {
            v16h af = load_afrag(A, SK, m0 + mt * 16, kt * 32, l);
            acc[mt] = __builtin_amdgcn_wmma_f32_16x16x32_f16(
                false, af, false, bf, (short)0, acc[mt], false, false);
        }
    }

    float bn = bia[n0 + (l & 15)];
    int mrow = ((l >> 4) << 3);
#pragma unroll
    for (int mt = 0; mt < 4; mt++) {
#pragma unroll
        for (int r = 0; r < 8; r++) {
            int m = m0 + mt * 16 + r + mrow;
            out[(size_t)m * GN + n0 + (l & 15)] = acc[mt][r] + bn;
        }
    }
}

// ---------------- LSTM scan (one block per direction) ----------------------
// xp: [T, 64, 400] f32 gate pre-activations (bias already folded in).
// whh: [400, 128] f16. 800 threads = 25 waves; wave n owns gate cols 16n..16n+15
// and keeps its 4 w_hh B-fragments in registers for the whole time loop.
// Per step: issue this step's xp loads into registers first (they fly during
// the WMMA phase and the LDS-only barrier), prefetch next step's xp to L2.
__global__ void lstm_scan_kernel(const float* __restrict__ xpF, const float* __restrict__ xpB,
                                 const _Float16* __restrict__ whhF, const _Float16* __restrict__ whhB,
                                 _Float16* __restrict__ h16out, int h16stride,  // null for layer 1
                                 float* __restrict__ hfinal,                    // null for layer 0
                                 int nstepsF, int nstepsB) {
    extern __shared__ char smem[];
    float*    g    = (float*)smem;                       // [64][400]
    _Float16* hbuf = (_Float16*)(smem + BB * GN * 4);    // [64][128]

    int dir = blockIdx.x;
    const float*    xp     = dir ? xpB : xpF;
    const _Float16* whh    = dir ? whhB : whhF;
    int             nsteps = dir ? nstepsB : nstepsF;
    int             colOff = dir * HV;
    int             tstride = dir ? -(BB * GN) : (BB * GN);

    int tid = threadIdx.x;      // 0..799
    int w   = tid >> 5;         // wave 0..24 -> N tile
    int l   = tid & 31;
    int n0  = w * 16;

    // per-thread elementwise slots: pairs p = tid + u*800 -> (b, j)
    int pb[8], pj[8];
#pragma unroll
    for (int u = 0; u < 8; u++) {
        int p = tid + u * 800;
        pb[u] = p / HV;
        pj[u] = p - pb[u] * HV;
    }

    // zero h state (pad cols 100..127 stay zero forever)
    for (int i = tid; i < BB * KH; i += 800) hbuf[i] = (_Float16)0.0f;

    // preload w_hh B fragments for all 4 K-steps (resident across time loop)
    v16h bfrag[4];
#pragma unroll
    for (int kt = 0; kt < 4; kt++) bfrag[kt] = load_bfrag(whh, KH, n0, kt * 32, l);

    float cst[8];
#pragma unroll
    for (int u = 0; u < 8; u++) cst[u] = 0.0f;

    wg_barrier_lds();

    for (int s = 0; s < nsteps; s++) {
        int t = dir ? (TT - 1 - s) : s;
        const float* xpt = xp + (size_t)t * BB * GN;

        // 1) issue this step's xp loads into registers (consumed after barrier)
        float xr[8][4];
#pragma unroll
        for (int u = 0; u < 8; u++) {
            const float* p = xpt + pb[u] * GN + pj[u];
            xr[u][0] = p[0];
            xr[u][1] = p[HV];
            xr[u][2] = p[2 * HV];
            xr[u][3] = p[3 * HV];
        }
        // 2) prefetch next step's xp slice HBM -> L2 (overlaps this step)
        if (s + 1 < nsteps) {
            const float* xpn = xpt + tstride;
#pragma unroll
            for (int u = 0; u < 8; u++) {
                const float* p = xpn + pb[u] * GN + pj[u];
                __builtin_prefetch(p, 0, 0);
                __builtin_prefetch(p + HV, 0, 0);
                __builtin_prefetch(p + 2 * HV, 0, 0);
                __builtin_prefetch(p + 3 * HV, 0, 0);
            }
        }

        // 3) g[0:64, n0:n0+16] = h @ whh^T  (K = 128 padded)
        v8f acc[4];
#pragma unroll
        for (int mt = 0; mt < 4; mt++) acc[mt] = (v8f)(0.0f);
#pragma unroll
        for (int kt = 0; kt < 4; kt++) {
#pragma unroll
            for (int mt = 0; mt < 4; mt++) {
                v16h af = load_afrag(hbuf, KH, mt * 16, kt * 32, l);
                acc[mt] = __builtin_amdgcn_wmma_f32_16x16x32_f16(
                    false, af, false, bfrag[kt], (short)0, acc[mt], false, false);
            }
        }
        int mrow = ((l >> 4) << 3);
#pragma unroll
        for (int mt = 0; mt < 4; mt++) {
#pragma unroll
            for (int r = 0; r < 8; r++) {
                int m = mt * 16 + r + mrow;
                g[m * GN + n0 + (l & 15)] = acc[mt][r];
            }
        }
        wg_barrier_lds();

        // 4) elementwise LSTM cell: 6400 (b,j) pairs, 8 per thread; c in regs
#pragma unroll
        for (int u = 0; u < 8; u++) {
            int b = pb[u], j = pj[u];
            int base = b * GN + j;
            float gi = g[base]          + xr[u][0];
            float gf = g[base + HV]     + xr[u][1];
            float gg = g[base + 2 * HV] + xr[u][2];
            float go = g[base + 3 * HV] + xr[u][3];
            float c  = sigm(gf) * cst[u] + sigm(gi) * tanhf(gg);
            cst[u]   = c;
            float h  = sigm(go) * tanhf(c);
            hbuf[b * KH + j] = (_Float16)h;
            if (h16out)
                h16out[(size_t)(t * BB + b) * h16stride + colOff + j] = (_Float16)h;
            if (hfinal && s == nsteps - 1)
                hfinal[b * (2 * HV) + colOff + j] = h;
        }
        wg_barrier_lds();
    }
}

// ---------------- final FC --------------------------------------------------
__global__ void fc_kernel(const float* __restrict__ hf, const float* __restrict__ fcw,
                          const float* __restrict__ fcb, float* __restrict__ out) {
    int tid = threadIdx.x;
    if (tid >= BB * 6) return;
    int b = tid / 6, c = tid - b * 6;
    float s = fcb[c];
    for (int j = 0; j < 2 * HV; j++) s += hf[b * 2 * HV + j] * fcw[c * 2 * HV + j];
    out[b * 6 + c] = s;
}

// ---------------- host ------------------------------------------------------

extern "C" void kernel_launch(void* const* d_in, const int* in_sizes, int n_in,
                              void* d_out, int out_size, void* d_ws, size_t ws_size,
                              hipStream_t stream) {
    (void)in_sizes; (void)n_in; (void)out_size; (void)ws_size;

    const int*   x     = (const int*)d_in[0];
    const float* emb   = (const float*)d_in[1];
    const float* fc_w  = (const float*)d_in[2];
    const float* fc_b  = (const float*)d_in[3];
    const float* wih[4] = {(const float*)d_in[4],  (const float*)d_in[8],
                           (const float*)d_in[12], (const float*)d_in[16]};
    const float* whh[4] = {(const float*)d_in[5],  (const float*)d_in[9],
                           (const float*)d_in[13], (const float*)d_in[17]};
    const float* bih[4] = {(const float*)d_in[6],  (const float*)d_in[10],
                           (const float*)d_in[14], (const float*)d_in[18]};
    const float* bhh[4] = {(const float*)d_in[7],  (const float*)d_in[11],
                           (const float*)d_in[15], (const float*)d_in[19]};

    char* ws = (char*)d_ws;
    size_t off = 0;
    auto alloc = [&](size_t bytes) { size_t o = off; off = (off + bytes + 255) & ~(size_t)255; return o; };

    _Float16* xe0    = (_Float16*)(ws + alloc((size_t)ROWS * DK0 * 2));
    _Float16* xe1    = (_Float16*)(ws + alloc((size_t)ROWS * DK1 * 2));
    float*    xpF    = (float*)   (ws + alloc((size_t)TT * BB * GN * 4)); // reused layer0/layer1
    float*    xpB    = (float*)   (ws + alloc((size_t)TT * BB * GN * 4));
    _Float16* wih16[4];
    wih16[0] = (_Float16*)(ws + alloc((size_t)GN * DK0 * 2));
    wih16[1] = (_Float16*)(ws + alloc((size_t)GN * DK0 * 2));
    wih16[2] = (_Float16*)(ws + alloc((size_t)GN * DK1 * 2));
    wih16[3] = (_Float16*)(ws + alloc((size_t)GN * DK1 * 2));
    _Float16* whh16[4];
    for (int i = 0; i < 4; i++) whh16[i] = (_Float16*)(ws + alloc((size_t)GN * KH * 2));
    float* bsum[4];
    for (int i = 0; i < 4; i++) bsum[i] = (float*)(ws + alloc((size_t)GN * 4));
    float* hfinal = (float*)(ws + alloc((size_t)BB * 2 * HV * 4));

    auto cvt = [&](const float* src, _Float16* dst, int rows, int sc, int dc) {
        int total = rows * dc;
        cvt_pad_kernel<<<(total + 255) / 256, 256, 0, stream>>>(src, dst, rows, sc, dc);
    };

    // 1) weight/bias prep
    cvt(wih[0], wih16[0], GN, 300, DK0);
    cvt(wih[1], wih16[1], GN, 300, DK0);
    cvt(wih[2], wih16[2], GN, 200, DK1);
    cvt(wih[3], wih16[3], GN, 200, DK1);
    for (int i = 0; i < 4; i++) {
        cvt(whh[i], whh16[i], GN, HV, KH);
        bias_sum_kernel<<<2, 256, 0, stream>>>(bih[i], bhh[i], bsum[i]);
    }

    // 2) embedding gather (f16, [T,B,320]) ; zero xe1 (for its 24 pad cols)
    embed_kernel<<<ROWS, DK0, 0, stream>>>(x, emb, xe0);
    hipMemsetAsync(xe1, 0, (size_t)ROWS * DK1 * 2, stream);

    // 3) layer-0 input projections (both directions)
    {
        dim3 grid(4, ROWS / 64, 2);
        proj_gemm_kernel<<<grid, 256, 0, stream>>>(xe0, wih16[0], wih16[1],
                                                   bsum[0], bsum[1], xpF, xpB, DK0);
    }

    // 4) layer-0 scan: writes h (f16) into xe1 cols [dir*100, dir*100+100)
    size_t scan_lds = (size_t)BB * GN * 4 + (size_t)BB * KH * 2;  // 118784 B
    lstm_scan_kernel<<<2, 800, scan_lds, stream>>>(xpF, xpB, whh16[0], whh16[1],
                                                   xe1, DK1, nullptr, TT, TT);

    // 5) layer-1 input projections
    {
        dim3 grid(4, ROWS / 64, 2);
        proj_gemm_kernel<<<grid, 256, 0, stream>>>(xe1, wih16[2], wih16[3],
                                                   bsum[2], bsum[3], xpF, xpB, DK1);
    }

    // 6) layer-1 scan: only final hidden needed; backward needs exactly 1 step
    lstm_scan_kernel<<<2, 800, scan_lds, stream>>>(xpF, xpB, whh16[2], whh16[3],
                                                   nullptr, 0, hfinal, TT, 1);

    // 7) classifier head
    fc_kernel<<<1, BB * 6, 0, stream>>>(hfinal, fc_w, fc_b, (float*)d_out);
}
